// GNN_71193377898818
// MI455X (gfx1250) — matrix operand order repaired and checked by reading, not compiled, not run on hardware
//
#include <hip/hip_runtime.h>

// Fully-fused EGNN forward for MI455X (gfx1250, wave32, WMMA).
// One kernel; all intermediates in LDS; every GEMM on v_wmma_f32_16x16x32_f16.
//
// Sizes: B=16384 graphs, N=5 nodes, NE=20 edges, H=60 (pad 64), L=4 layers.
// Block = 128 threads (4 waves), 64 graphs/block, 256 blocks.
// Wave w owns graphs [w*16, w*16+16): 320 edge rows = 20 M-tiles,
// 80 node rows = 5 M-tiles; a wave's edges touch only that wave's nodes, and
// node i's segment-sum is the sum of edge rows [4i,4i+4) (static edge list).
//
// A-fragment trick: 16-bit A element e -> K = (e/8)*16 + (lane/16)*8 + e%8,
// i.e. two contiguous 16-byte runs -> every A fragment is 2x ds_load_b128.
// Bias trick: activation column 60 (padding) holds constant 1.0 and each
// bias vector is staged as weight row K=60 -> all MLPs start from an
// inline-0 WMMA accumulator.
// ef trick: the 2 edge features ride in dst-half padding columns K=125/126,
// which map exclusively to elements 13/14 of the kt=3 fragment in lanes
// 16..31 -> patched with two v_cndmask_b16; ew1 rows 120/121 staged there.

typedef __attribute__((ext_vector_type(16))) _Float16 v16h;
typedef __attribute__((ext_vector_type(8)))  _Float16 v8h;
typedef __attribute__((ext_vector_type(8)))  float    v8f;

#define TPB   128
#define GPB   64          // graphs per block
#define NPB   320         // nodes per block  (GPB*5)
#define EPB   1280        // edges per block  (GPB*20)

// ---- dynamic LDS partition (bytes) ----
#define OFF_H     0                 // h        : [320][64] f16 (col60=1, 61..63=0)
#define OFF_AGG   40960             // agg / d1 : [320][64] f16 (cols 60..63 = 0)
#define OFF_EF    81920             // ef       : [1280][2] f16 = 5120
#define OFF_W1E   87040             // edge W1 frags [4][4][32][16] f16 = 16384
#define OFF_W2E   103424            // edge W2 frags [2][4][32][16] f16 = 8192
#define OFF_W1N   111616            // node W1 frags                    = 16384
#define OFF_W2N   119808            // node W2 frags                    = 8192
#define OFF_DW2   128000            // dec_w2 (60*3) + dec_b2 (3) f32   = 1024
#define OFF_SCR   129024            // per-wave scratch 4*[16][64] f16  = 8192
#define SMEM_BYTES 137216

__device__ __forceinline__ float silu_f(float x) {
  return x * (1.0f / (1.0f + __expf(-x)));
}

__device__ __forceinline__ v8f wmma_f16(v16h a, v16h b, v8f c) {
  // (neg_a, A, neg_b, B, c_mod, C, reuse_a, reuse_b)
  return __builtin_amdgcn_wmma_f32_16x16x32_f16(false, a, false, b, (short)0, c,
                                                false, false);
}

// B fragments pre-swizzled in LDS: 16 contiguous halves per lane per fragment
// (lane holds col N=lane%16; element e maps to K = (e/8)*16 + (lane/16)*8 + e%8).
__device__ __forceinline__ v16h load_bfrag(const _Float16* base, int kt, int nt,
                                           int lane) {
  return *(const v16h*)(base + (((kt * 4 + nt) * 32 + lane) << 4));
}

// A fragment from one contiguous 64-half LDS row: two 16-byte runs.
__device__ __forceinline__ v16h load_afrag(const _Float16* row, int kb) {
  v8h lo = *(const v8h*)(row + kb);
  v8h hi = *(const v8h*)(row + kb + 16);
  return __builtin_shufflevector(lo, hi, 0, 1, 2, 3, 4, 5, 6, 7,
                                 8, 9, 10, 11, 12, 13, 14, 15);
}

// Stage [.][60] f32 weight (rows 0..59 -> K 0..59, rows 60..119 -> K 64..123)
// into fragment-major f16; bias staged at K=60 (activation col60 == 1.0);
// if efrows, rows 120/121 staged at K=125/126 (edge-feature columns).
__device__ __forceinline__ void stage_wfrag128(_Float16* dst, const float* w,
                                               const float* bias, int efrows,
                                               int tid) {
  for (int s = tid; s < 8192; s += TPB) {
    int e = s & 15, lane = (s >> 4) & 31, nt = (s >> 9) & 3, kt = s >> 11;
    int k = kt * 32 + ((e >> 3) << 4) + ((lane >> 4) << 3) + (e & 7);      // B-fragment K mapping
    int n = nt * 16 + (lane & 15);
    int kr = k & 63, half = k >> 6;
    float v = 0.0f;
    if (n < 60) {
      if (kr < 60)                 v = w[(half * 60 + kr) * 60 + n];
      else if (kr == 60 && !half)  v = bias[n];
      else if (efrows && half && (kr == 61 || kr == 62))
        v = w[(120 + (kr - 61)) * 60 + n];         // ew1 rows 120,121
    }
    dst[s] = (_Float16)v;
  }
}

// Stage [60][60] f32 weight into fragment-major f16 (K pad 64, bias at K=60).
__device__ __forceinline__ void stage_wfrag64(_Float16* dst, const float* w,
                                              const float* bias, int tid) {
  for (int s = tid; s < 4096; s += TPB) {
    int e = s & 15, lane = (s >> 4) & 31, nt = (s >> 9) & 3, kt = s >> 11;
    int k = kt * 32 + ((e >> 3) << 4) + ((lane >> 4) << 3) + (e & 7);
    int n = nt * 16 + (lane & 15);
    float v = 0.0f;
    if (n < 60) {
      if (k < 60)       v = w[k * 60 + n];
      else if (k == 60) v = bias[n];
    }
    dst[s] = (_Float16)v;
  }
}

__global__ __launch_bounds__(TPB) void egnn_fused(
    const float* __restrict__ nf,    const float* __restrict__ efeat,
    const float* __restrict__ emb_w, const float* __restrict__ emb_bv,
    const float* __restrict__ ew1,   const float* __restrict__ eb1,
    const float* __restrict__ ew2,   const float* __restrict__ eb2,
    const float* __restrict__ nw1,   const float* __restrict__ nb1,
    const float* __restrict__ nw2,   const float* __restrict__ nb2,
    const float* __restrict__ dw1,   const float* __restrict__ db1,
    const float* __restrict__ dw2,   const float* __restrict__ db2,
    float* __restrict__ out)
{
  extern __shared__ char smem[];
  _Float16* sh_h   = (_Float16*)(smem + OFF_H);
  _Float16* sh_agg = (_Float16*)(smem + OFF_AGG);
  _Float16* sh_ef  = (_Float16*)(smem + OFF_EF);
  _Float16* sh_w1e = (_Float16*)(smem + OFF_W1E);
  _Float16* sh_w2e = (_Float16*)(smem + OFF_W2E);
  _Float16* sh_w1n = (_Float16*)(smem + OFF_W1N);
  _Float16* sh_w2n = (_Float16*)(smem + OFF_W2N);
  float*    sh_dw2 = (float*)   (smem + OFF_DW2);
  _Float16* sh_scr = (_Float16*)(smem + OFF_SCR);

  const int tid  = threadIdx.x;
  const int wave = tid >> 5;
  const int lane = tid & 31;
  const int ln16 = lane & 15;
  const int hiK  = (lane >> 4);               // 0: K-run 0..7, 1: K-run 8..15
  const int oK   = hiK << 3;
  const int blk  = blockIdx.x;
  _Float16* scr  = sh_scr + wave * 16 * 64;   // per-wave C->A transpose scratch
  const _Float16 one16 = (_Float16)1.0f;

  // ---- embedding (6->60, tiny: scalar) + edge-feature staging ----
  for (int s = tid; s < NPB * 64; s += TPB) {
    int col = s & 63, row = s >> 6;
    float acc = 0.0f;
    if (col < 60) {
      acc = emb_bv[col];
      int grow = blk * NPB + row;               // global (b*5+n)
#pragma unroll
      for (int k = 0; k < 6; ++k)               // x[row][k] = nf[b][n][k%3][k/3]
        acc += nf[(grow * 3 + (k % 3)) * 2 + (k / 3)] * emb_w[k * 60 + col];
    } else if (col == 60) {
      acc = 1.0f;                               // bias hook column
    }
    sh_h[s] = (_Float16)acc;
  }
  for (int s = tid; s < EPB * 2; s += TPB) {
    int c = s & 1, le = s >> 1;
    int g = le / 20, e = le % 20, i = e >> 2, jp = e & 3;
    int j = jp + (jp >= i ? 1 : 0);
    int gb = blk * GPB + g;
    sh_ef[s] = (_Float16)efeat[((gb * 5 + i) * 5 + j) * 2 + c];
  }
  __syncthreads();

  // =================== 4 GNN layers ===================
  for (int l = 0; l < 4; ++l) {
    stage_wfrag128(sh_w1e, ew1 + l * 122 * 60, eb1 + l * 60, 1, tid);
    stage_wfrag64 (sh_w2e, ew2 + l * 60 * 60,  eb2 + l * 60, tid);
    stage_wfrag128(sh_w1n, nw1 + l * 120 * 60, nb1 + l * 60, 0, tid);
    stage_wfrag64 (sh_w2n, nw2 + l * 60 * 60,  nb2 + l * 60, tid);
    if (tid == 0 && l < 3) {                           // global_prefetch next layer
      __builtin_prefetch(ew1 + (l + 1) * 122 * 60, 0, 1);
      __builtin_prefetch(ew2 + (l + 1) * 60 * 60, 0, 1);
      __builtin_prefetch(nw1 + (l + 1) * 120 * 60, 0, 1);
      __builtin_prefetch(nw2 + (l + 1) * 60 * 60, 0, 1);
    }
    __syncthreads();

    // ---------- edge phase: 20 M-tiles of 16 edges per wave ----------
#pragma unroll 1
    for (int t = 0; t < 20; ++t) {
      const int base = wave * 320 + t * 16;     // local edge row base
      const int r    = base + ln16;             // this lane's A-row (edge)
      const int g    = r / 20, eid = r % 20;
      const int si   = eid >> 2, jp = eid & 3;
      const _Float16* rowS = sh_h + (g * 5 + si) * 64;
      const _Float16* rowD = sh_h + (g * 5 + jp + (jp >= si ? 1 : 0)) * 64;

      // A = [h_src(64) | h_dst(64)]: kt 0,1 from src row, kt 2,3 from dst row
      v16h A[4];
#pragma unroll
      for (int kt = 0; kt < 4; ++kt)
        A[kt] = load_afrag((kt < 2) ? rowS : rowD, ((kt & 1) << 5) + oK);
      // patch ef into K=125/126 (kt=3 elements 13/14, lanes 16..31 only)
      {
        _Float16 e0 = sh_ef[r * 2 + 0], e1 = sh_ef[r * 2 + 1];
        A[3][13] = hiK ? e0 : A[3][13];
        A[3][14] = hiK ? e1 : A[3][14];
      }

      // MLP1: silu(A@W1)  (bias/ef inside W1 rows 60/125/126)   (16 WMMAs)
      v8f C1[4] = {};
#pragma unroll
      for (int kt = 0; kt < 4; ++kt)            // kt outer: independent accs
#pragma unroll
        for (int nt = 0; nt < 4; ++nt)
          C1[nt] = wmma_f16(A[kt], load_bfrag(sh_w1e, kt, nt, lane), C1[nt]);

      // silu + C->A transpose through per-wave LDS scratch (col60 := 1.0)
#pragma unroll
      for (int nt = 0; nt < 4; ++nt) {
        int n = nt * 16 + ln16;
#pragma unroll
        for (int r8 = 0; r8 < 8; ++r8) {
          _Float16 v = (_Float16)silu_f(C1[nt][r8]);
          if (nt == 3) v = (ln16 == 12) ? one16 : v;   // n==60 -> 1.0
          scr[(oK + r8) * 64 + n] = v;
        }
      }
      asm volatile("s_wait_dscnt 0" ::: "memory");
      const _Float16* sp = scr + ln16 * 64;
      v16h A2[2];
#pragma unroll
      for (int kt = 0; kt < 2; ++kt) A2[kt] = load_afrag(sp, (kt << 5) + oK);

      // MLP2: silu(.@W2) (bias in W2 row60)  (8 WMMAs), then segment-sum
      v8f C2[4] = {};
#pragma unroll
      for (int kt = 0; kt < 2; ++kt)
#pragma unroll
        for (int nt = 0; nt < 4; ++nt)
          C2[nt] = wmma_f16(A2[kt], load_bfrag(sh_w2e, kt, nt, lane), C2[nt]);
#pragma unroll
      for (int nt = 0; nt < 4; ++nt) {
        v8f c = C2[nt];
#pragma unroll
        for (int r8 = 0; r8 < 8; ++r8) c[r8] = silu_f(c[r8]);
        // C layout: VGPR r8 holds row M = r8 + 8*(lane/16); node = edge_row/4
        float g0 = c[0] + c[1] + c[2] + c[3];
        float g1 = c[4] + c[5] + c[6] + c[7];
        int n = nt * 16 + ln16;
        int nodeb = (base >> 2) + hiK * 2;
        sh_agg[(nodeb + 0) * 64 + n] = (_Float16)g0;
        sh_agg[(nodeb + 1) * 64 + n] = (_Float16)g1;
      }
    }
    asm volatile("s_wait_dscnt 0" ::: "memory");  // same-wave agg RAW

    // ---------- node phase: 5 M-tiles of 16 nodes per wave ----------
#pragma unroll 1
    for (int t = 0; t < 5; ++t) {
      const int base = wave * 80 + t * 16;
      const int row  = base + ln16;
      const _Float16* rowH = sh_h   + row * 64;
      const _Float16* rowA = sh_agg + row * 64;
      v16h A[4];                                  // [h(64) | agg(64)]
#pragma unroll
      for (int kt = 0; kt < 4; ++kt)
        A[kt] = load_afrag((kt < 2) ? rowH : rowA, ((kt & 1) << 5) + oK);
      v8f C1[4] = {};
#pragma unroll
      for (int kt = 0; kt < 4; ++kt)
#pragma unroll
        for (int nt = 0; nt < 4; ++nt)
          C1[nt] = wmma_f16(A[kt], load_bfrag(sh_w1n, kt, nt, lane), C1[nt]);
#pragma unroll
      for (int nt = 0; nt < 4; ++nt) {
        int n = nt * 16 + ln16;
#pragma unroll
        for (int r8 = 0; r8 < 8; ++r8) {
          _Float16 v = (_Float16)silu_f(C1[nt][r8]);
          if (nt == 3) v = (ln16 == 12) ? one16 : v;   // n==60 -> 1.0
          scr[(oK + r8) * 64 + n] = v;
        }
      }
      asm volatile("s_wait_dscnt 0" ::: "memory");
      const _Float16* sp = scr + ln16 * 64;
      v16h A2[2];
#pragma unroll
      for (int kt = 0; kt < 2; ++kt) A2[kt] = load_afrag(sp, (kt << 5) + oK);
      v8f C2[4] = {};
#pragma unroll
      for (int kt = 0; kt < 2; ++kt)              // 60->60, no activation
#pragma unroll
        for (int nt = 0; nt < 4; ++nt)
          C2[nt] = wmma_f16(A2[kt], load_bfrag(sh_w2n, kt, nt, lane), C2[nt]);
#pragma unroll
      for (int nt = 0; nt < 4; ++nt) {
        int n = nt * 16 + ln16;
#pragma unroll
        for (int r8 = 0; r8 < 8; ++r8) {
          _Float16 v = (_Float16)C2[nt][r8];
          if (nt == 3) v = (ln16 == 12) ? one16 : v;   // keep h[.][60] == 1.0
          sh_h[(base + oK + r8) * 64 + n] = v;
        }
      }
      asm volatile("s_wait_dscnt 0" ::: "memory");
    }
    __syncthreads();   // all waves done before next layer's weight restage
  }

  // =================== decoder ===================
  stage_wfrag64(sh_w2e, dw1, db1, tid);           // reuse edge-W2 frag buffer
  for (int s = tid; s < 183; s += TPB)
    sh_dw2[s] = (s < 180) ? dw2[s] : db2[s - 180];
  __syncthreads();

#pragma unroll 1
  for (int t = 0; t < 5; ++t) {                   // d1 = silu(h @ dw1 + b1)
    const int base = wave * 80 + t * 16;
    const _Float16* rowH = sh_h + (base + ln16) * 64;
    v16h A2[2];
#pragma unroll
    for (int kt = 0; kt < 2; ++kt) A2[kt] = load_afrag(rowH, (kt << 5) + oK);
    v8f C2[4] = {};
#pragma unroll
    for (int kt = 0; kt < 2; ++kt)
#pragma unroll
      for (int nt = 0; nt < 4; ++nt)
        C2[nt] = wmma_f16(A2[kt], load_bfrag(sh_w2e, kt, nt, lane), C2[nt]);
#pragma unroll
    for (int nt = 0; nt < 4; ++nt) {
      int n = nt * 16 + ln16;
#pragma unroll
      for (int r8 = 0; r8 < 8; ++r8)
        sh_agg[(base + oK + r8) * 64 + n] = (_Float16)silu_f(C2[nt][r8]);  // d1
    }
  }
  asm volatile("s_wait_dscnt 0" ::: "memory");

  // final 60->3 GEMV (tiny): scalar per node row, then store to global
  for (int r = lane; r < 80; r += 32) {
    int row = wave * 80 + r;
    float a0 = sh_dw2[180], a1 = sh_dw2[181], a2 = sh_dw2[182];
    for (int k = 0; k < 60; ++k) {
      float v = (float)sh_agg[row * 64 + k];
      a0 += v * sh_dw2[k * 3 + 0];
      a1 += v * sh_dw2[k * 3 + 1];
      a2 += v * sh_dw2[k * 3 + 2];
    }
    int grow = blk * NPB + row;
    out[grow * 3 + 0] = a0;
    out[grow * 3 + 1] = a1;
    out[grow * 3 + 2] = a2;
  }
}

extern "C" void kernel_launch(void* const* d_in, const int* in_sizes, int n_in,
                              void* d_out, int out_size, void* d_ws, size_t ws_size,
                              hipStream_t stream) {
  (void)in_sizes; (void)n_in; (void)out_size; (void)d_ws; (void)ws_size;
  const float* nf    = (const float*)d_in[0];
  const float* efeat = (const float*)d_in[1];
  // d_in[2] = edge_idx (int64): static fully-connected pattern, derived on-chip.
  const float* emb_w = (const float*)d_in[3];
  const float* emb_b = (const float*)d_in[4];
  const float* ew1   = (const float*)d_in[5];
  const float* eb1   = (const float*)d_in[6];
  const float* ew2   = (const float*)d_in[7];
  const float* eb2   = (const float*)d_in[8];
  const float* nw1   = (const float*)d_in[9];
  const float* nb1   = (const float*)d_in[10];
  const float* nw2   = (const float*)d_in[11];
  const float* nb2   = (const float*)d_in[12];
  const float* dw1   = (const float*)d_in[13];
  const float* db1   = (const float*)d_in[14];
  const float* dw2   = (const float*)d_in[15];
  const float* db2   = (const float*)d_in[16];
  float* out = (float*)d_out;

  (void)hipFuncSetAttribute((const void*)egnn_fused,
                            hipFuncAttributeMaxDynamicSharedMemorySize, SMEM_BYTES);
  // 16384 graphs / 64 per block = 256 blocks, 128 threads (4 wave32s) each.
  egnn_fused<<<256, TPB, SMEM_BYTES, stream>>>(nf, efeat, emb_w, emb_b,
                                               ew1, eb1, ew2, eb2,
                                               nw1, nb1, nw2, nb2,
                                               dw1, db1, dw2, db2, out);
}
